// GatedLinearAttention_81183471829678
// MI455X (gfx1250) — matrix operand order repaired and checked by reading, not compiled, not run on hardware
//
#include <hip/hip_runtime.h>

// ---------------------------------------------------------------------------
// GLA forward for MI455X (gfx1250, wave32, WMMA).
// B=2, L=4096, HID=2048, H=16, D=128, C=64, LOWR=16, GLN=16, EPS=1e-5
// ---------------------------------------------------------------------------

typedef __bf16 bf16_t;
typedef __bf16 v16bf __attribute__((ext_vector_type(16)));
typedef float  v8f   __attribute__((ext_vector_type(8)));

#define BB     2
#define LL     4096
#define HIDD   2048
#define HH     16
#define DD     128
#define CC     64
#define NC     64
#define MM     8192          // B*L
#define BH     32            // B*H
#define EPSF   1e-5f
#define SCALE  0.08838834764831845f   // D^-0.5

union BFrag { v16bf v; uint4 q[2]; };

// A-matrix fragment (16x32 bf16) from a row-major, K-contiguous tile.
// lane<16: K 0..7 then 16..23 of row (row0+lane); lane>=16: K 8..15 then 24..31.
__device__ __forceinline__ v16bf ld_fragA(const bf16_t* tile, int row0, int stride,
                                          int k0, int lane) {
    int l15 = lane & 15;
    int h   = (lane >> 4) & 1;
    const bf16_t* p = tile + (row0 + l15) * stride + k0 + h * 8;
    BFrag u;
    u.q[0] = *(const uint4*)(p);
    u.q[1] = *(const uint4*)(p + 16);
    return u.v;
}

// B-matrix fragment (32x16 bf16) from a TRANSPOSED tile (N rows, K contiguous).
// lane<16: K 0..7 then 16..23 of column (row0+lane); lane>=16: K 8..15 then 24..31.
__device__ __forceinline__ v16bf ld_fragB(const bf16_t* tileT, int row0, int stride,
                                          int k0, int lane) {
    int l15 = lane & 15;
    int h   = (lane >> 4) & 1;
    const bf16_t* p = tileT + (row0 + l15) * stride + k0 + h * 8;
    BFrag u;
    u.q[0] = *(const uint4*)(p);
    u.q[1] = *(const uint4*)(p + 16);
    return u.v;
}

__device__ __forceinline__ v8f wmma_bf16(v16bf a, v16bf b, v8f c) {
    return __builtin_amdgcn_wmma_f32_16x16x32_bf16(false, a, false, b,
                                                   (short)0, c, false, false);
}

// ---------------------------------------------------------------------------
// Elementwise fp32 -> bf16
// ---------------------------------------------------------------------------
__global__ void f32_to_bf16_kernel(const float* __restrict__ in,
                                   bf16_t* __restrict__ out, size_t n) {
    size_t i = (size_t)blockIdx.x * blockDim.x + threadIdx.x;
    if (i < n) out[i] = (bf16_t)in[i];
}

// Weight transpose + convert: out[n*K + k] = (bf16) in[k*N + n]
__global__ void transpose_to_bf16_kernel(const float* __restrict__ in,
                                         bf16_t* __restrict__ out, int K, int N) {
    size_t idx = (size_t)blockIdx.x * blockDim.x + threadIdx.x;
    size_t total = (size_t)K * N;
    if (idx >= total) return;
    int n = (int)(idx / K);
    int k = (int)(idx % K);
    out[idx] = (bf16_t)in[(size_t)k * N + n];
}

// ---------------------------------------------------------------------------
// Tiled bf16 WMMA GEMM:  C(fp32, MxN) = A(bf16, MxK, row-major) * Bt^T
// Bt is B stored transposed: (N rows, K contiguous).
// Block = 256 threads (8 waves), tile 128x128, K-step 32.
// Double-buffered LDS (ping-pong), global loads issued one K-step ahead so
// VMEM latency overlaps WMMA compute; one barrier per K-step.
// Wave grid 4(M) x 2(N); each wave computes 32x64 = 2x4 wmma tiles.
// ---------------------------------------------------------------------------
__global__ __launch_bounds__(256)
void gemm_bf16_kernel(const bf16_t* __restrict__ A, const bf16_t* __restrict__ Bt,
                      float* __restrict__ C, int Mdim, int Ndim, int Kdim) {
    __shared__ __align__(16) bf16_t As[2][128 * 32];
    __shared__ __align__(16) bf16_t Bs[2][128 * 32];

    int tid  = threadIdx.x;
    int lane = tid & 31;
    int w    = tid >> 5;
    int mw   = w >> 1;     // 0..3
    int nw   = w & 1;      // 0..1
    int m0   = blockIdx.y * 128;
    int n0   = blockIdx.x * 128;

    // Each thread stages two 16-byte chunks per matrix per K-step.
    int c0 = tid, c1 = tid + 256;
    int r0 = c0 >> 2, q0 = c0 & 3;
    int r1 = c1 >> 2, q1 = c1 & 3;
    const bf16_t* pA0 = A  + (size_t)(m0 + r0) * Kdim + q0 * 8;
    const bf16_t* pA1 = A  + (size_t)(m0 + r1) * Kdim + q1 * 8;
    const bf16_t* pB0 = Bt + (size_t)(n0 + r0) * Kdim + q0 * 8;
    const bf16_t* pB1 = Bt + (size_t)(n0 + r1) * Kdim + q1 * 8;

    uint4 ra0, ra1, rb0, rb1;
    auto ldglb = [&](int kt) {
        size_t off = (size_t)kt * 32;
        ra0 = *(const uint4*)(pA0 + off);
        ra1 = *(const uint4*)(pA1 + off);
        rb0 = *(const uint4*)(pB0 + off);
        rb1 = *(const uint4*)(pB1 + off);
    };
    auto stlds = [&](int buf) {
        *(uint4*)(&As[buf][r0 * 32 + q0 * 8]) = ra0;
        *(uint4*)(&As[buf][r1 * 32 + q1 * 8]) = ra1;
        *(uint4*)(&Bs[buf][r0 * 32 + q0 * 8]) = rb0;
        *(uint4*)(&Bs[buf][r1 * 32 + q1 * 8]) = rb1;
    };

    v8f zero = {};
    v8f acc[2][4];
    for (int i = 0; i < 2; i++)
        for (int j = 0; j < 4; j++) acc[i][j] = zero;

    int KT = Kdim >> 5;
    ldglb(0);
    stlds(0);
    if (KT > 1) ldglb(1);
    __syncthreads();

    for (int kt = 0; kt < KT; kt++) {
        int cur = kt & 1;
        if (kt + 1 < KT) stlds(cur ^ 1);     // fill idle buffer (prefetched regs)
        if (kt + 2 < KT) ldglb(kt + 2);      // issue next global loads early

        v16bf af[2], bfr[4];
        for (int i = 0; i < 2; i++) af[i]  = ld_fragA(As[cur], mw * 32 + i * 16, 32, 0, lane);
        for (int j = 0; j < 4; j++) bfr[j] = ld_fragB(Bs[cur], nw * 64 + j * 16, 32, 0, lane);
        for (int i = 0; i < 2; i++)
            for (int j = 0; j < 4; j++)
                acc[i][j] = wmma_bf16(af[i], bfr[j], acc[i][j]);
        __syncthreads();
    }

    int l15 = lane & 15, h = (lane >> 4) & 1;
    for (int i = 0; i < 2; i++)
        for (int j = 0; j < 4; j++) {
            int col = n0 + nw * 64 + j * 16 + l15;
            for (int r = 0; r < 8; r++) {
                int row = m0 + mw * 32 + i * 16 + r + h * 8;
                C[(size_t)row * Ndim + col] = acc[i][j][r];
            }
        }
}

// ---------------------------------------------------------------------------
// Low-rank gate: gk = log_sigmoid(x @ Wg1 @ Wg2 + bg2) / GLN  (one row/block)
// ---------------------------------------------------------------------------
__global__ __launch_bounds__(256)
void lowrank_gate_kernel(const float* __restrict__ x, const float* __restrict__ Wg1,
                         const float* __restrict__ Wg2, const float* __restrict__ bg2,
                         float* __restrict__ gk) {
    __shared__ float partial[8][16];
    __shared__ float t[16];
    int tid = threadIdx.x, lane = tid & 31, w = tid >> 5;
    size_t row = blockIdx.x;

    float acc[16];
    for (int j = 0; j < 16; j++) acc[j] = 0.0f;
    for (int k = tid; k < HIDD; k += 256) {
        float xv = x[row * HIDD + k];
        const float* wr = Wg1 + (size_t)k * 16;
        for (int j = 0; j < 16; j++) acc[j] += xv * wr[j];
    }
    for (int j = 0; j < 16; j++) {
        float s = acc[j];
        for (int off = 16; off; off >>= 1) s += __shfl_xor(s, off, 32);
        if (lane == 0) partial[w][j] = s;
    }
    __syncthreads();
    if (tid < 16) {
        float s = 0.0f;
        for (int ww = 0; ww < 8; ww++) s += partial[ww][tid];
        t[tid] = s;
    }
    __syncthreads();
    for (int n = tid; n < HIDD; n += 256) {
        float s = bg2[n];
        for (int j = 0; j < 16; j++) s += t[j] * Wg2[(size_t)j * HIDD + n];
        float ls = (s >= 0.0f) ? -log1pf(__expf(-s)) : (s - log1pf(__expf(s)));
        gk[row * HIDD + n] = ls * 0.0625f;   // /GLN
    }
}

// ---------------------------------------------------------------------------
// Per-chunk cumulative sum of gk along l (within each chunk).
// ---------------------------------------------------------------------------
__global__ __launch_bounds__(256)
void cumsum_kernel(const float* __restrict__ gk, float* __restrict__ Gc) {
    int blk = blockIdx.x;
    int b = blk >> 6, c = blk & 63;
    size_t rowbase = (size_t)b * LL + (size_t)c * CC;
    for (int col = threadIdx.x; col < HIDD; col += 256) {
        float run = 0.0f;
        for (int l = 0; l < CC; l++) {
            size_t o = (rowbase + l) * HIDD + col;
            run += gk[o];
            Gc[o] = run;
        }
    }
}

// ---------------------------------------------------------------------------
// State recurrence: for each (b,h,d) row of S, scan chunks sequentially.
// Stores pre-chunk states Sst[c] and final S into d_out tail.
// ---------------------------------------------------------------------------
__global__ __launch_bounds__(256)
void state_scan_kernel(const float* __restrict__ k, const float* __restrict__ v,
                       const float* __restrict__ Gc, float* __restrict__ Sst,
                       float* __restrict__ Sfinal) {
    __shared__ float Vs[CC * DD];   // 32 KB
    int tid = threadIdx.x, lane = tid & 31, w = tid >> 5;
    int blk  = blockIdx.x;
    int bh   = blk >> 4;
    int dblk = blk & 15;
    int d    = dblk * 8 + w;
    int b    = bh >> 4, h = bh & 15;
    int colbase = h * DD;

    float s0 = 0.f, s1 = 0.f, s2 = 0.f, s3 = 0.f;
    for (int c = 0; c < NC; c++) {
        size_t rowbase = (size_t)b * LL + (size_t)c * CC;
        for (int idx = tid; idx < CC * DD; idx += 256)
            Vs[idx] = v[(rowbase + (idx >> 7)) * HIDD + colbase + (idx & 127)];
        __syncthreads();

        float* Sd = Sst + (((size_t)c * BH + bh) * DD + d) * DD;
        Sd[lane] = s0; Sd[lane + 32] = s1; Sd[lane + 64] = s2; Sd[lane + 96] = s3;

        float glast = Gc[(rowbase + (CC - 1)) * HIDD + colbase + d];
        float decay = __expf(glast);
        s0 *= decay; s1 *= decay; s2 *= decay; s3 *= decay;

        for (int l = 0; l < CC; l++) {
            size_t o  = (rowbase + l) * HIDD + colbase + d;
            float wgt = k[o] * __expf(glast - Gc[o]);
            s0 += wgt * Vs[l * DD + lane];
            s1 += wgt * Vs[l * DD + lane + 32];
            s2 += wgt * Vs[l * DD + lane + 64];
            s3 += wgt * Vs[l * DD + lane + 96];
        }
        __syncthreads();
    }
    float* Sf = Sfinal + ((size_t)bh * DD + d) * DD;
    Sf[lane] = s0; Sf[lane + 32] = s1; Sf[lane + 64] = s2; Sf[lane + 96] = s3;
}

// ---------------------------------------------------------------------------
// Per-chunk output: o = qg@S + tril(Qh@Kh^T)@V, then RMS-norm * norm_w * silu(g).
// All matmuls via bf16 WMMA; output written directly as bf16 for the final GEMM.
// grid = (NC, B*H), block 256 (8 waves, 4x2 wave grid).
// ---------------------------------------------------------------------------
__global__ __launch_bounds__(256)
void chunk_output_kernel(const float* __restrict__ q, const float* __restrict__ k,
                         const float* __restrict__ v, const float* __restrict__ Gc,
                         const float* __restrict__ Sst, const float* __restrict__ g,
                         const float* __restrict__ norm_w, bf16_t* __restrict__ O) {
    __shared__ __align__(16) bf16_t sA[CC * DD];   // 16 KB: Qh -> V^T -> qg
    __shared__ __align__(16) bf16_t sB[DD * DD];   // 32 KB: Kh(+Abf) -> S^T -> obuf
    __shared__ float sG0[DD];

    int tid = threadIdx.x, lane = tid & 31, w = tid >> 5;
    int cidx = blockIdx.x;
    int bh   = blockIdx.y;
    int b = bh >> 4, h = bh & 15;
    size_t rowbase = (size_t)b * LL + (size_t)cidx * CC;
    int colbase = h * DD;

    if (tid < DD) sG0[tid] = Gc[rowbase * HIDD + colbase + tid];
    __syncthreads();

    // Stage Qh (sA) and Kh (sB rows 0..63), both row-major K(d)-contiguous.
    for (int idx = tid; idx < CC * DD; idx += 256) {
        int l = idx >> 7, d = idx & 127;
        size_t o = (rowbase + l) * HIDD + colbase + d;
        float gc = Gc[o], g0 = sG0[d];
        sA[idx] = (bf16_t)(q[o] * SCALE * __expf(gc - g0));
        sB[idx] = (bf16_t)(k[o] * __expf(g0 - gc));
    }
    __syncthreads();

    v8f zero = {};
    // ---- A = Qh @ Kh^T (64x64, K=128); wave tile 16x32 = 1x2 wmma tiles ----
    int mw = w >> 1, nw = w & 1;
    v8f accA[2] = {zero, zero};
    for (int k0 = 0; k0 < DD; k0 += 32) {
        v16bf af = ld_fragA(sA, mw * 16, DD, k0, lane);
        for (int j = 0; j < 2; j++) {
            v16bf bfr = ld_fragB(sB, nw * 32 + j * 16, DD, k0, lane);
            accA[j] = wmma_bf16(af, bfr, accA[j]);
        }
    }
    // Mask (keep s<=l) and store A bf16 into sB rows 64..127 (disjoint region).
    bf16_t* Abf = sB + CC * DD;   // 64x64, stride 64
    {
        int l15 = lane & 15, hh = (lane >> 4) & 1;
        for (int j = 0; j < 2; j++) {
            int s = nw * 32 + j * 16 + l15;
            for (int r = 0; r < 8; r++) {
                int l = mw * 16 + r + hh * 8;
                float val = (s <= l) ? accA[j][r] : 0.0f;
                Abf[l * CC + s] = (bf16_t)val;
            }
        }
    }
    __syncthreads();

    // Stage V^T into sA (128 rows of v-index, 64 s-contiguous).
    for (int idx = tid; idx < CC * DD; idx += 256) {
        int dv = idx >> 6, l = idx & 63;
        sA[idx] = (bf16_t)(v[(rowbase + l) * HIDD + colbase + dv]);
    }
    __syncthreads();

    // ---- o_intra = A @ V (64x128, K=64); wave tile 16x64 = 1x4 wmma tiles ----
    v8f acco[4] = {zero, zero, zero, zero};
    for (int k0 = 0; k0 < CC; k0 += 32) {
        v16bf af = ld_fragA(Abf, mw * 16, CC, k0, lane);
        for (int j = 0; j < 4; j++) {
            v16bf bfr = ld_fragB(sA, nw * 64 + j * 16, CC, k0, lane);
            acco[j] = wmma_bf16(af, bfr, acco[j]);
        }
    }
    __syncthreads();

    // Stage qg into sA and S^T into sB.
    for (int idx = tid; idx < CC * DD; idx += 256) {
        int l = idx >> 7, d = idx & 127;
        size_t o = (rowbase + l) * HIDD + colbase + d;
        sA[idx] = (bf16_t)(q[o] * SCALE * __expf(Gc[o]));
    }
    const float* Sc = Sst + ((size_t)cidx * BH + bh) * DD * DD;
    for (int idx = tid; idx < DD * DD; idx += 256) {
        int vv = idx >> 7, dd = idx & 127;
        sB[idx] = (bf16_t)Sc[(size_t)dd * DD + vv];   // S^T[v][d] = S[d][v]
    }
    __syncthreads();

    // ---- o_inter = qg @ S (64x128, K=128), accumulate into acco ----
    for (int k0 = 0; k0 < DD; k0 += 32) {
        v16bf af = ld_fragA(sA, mw * 16, DD, k0, lane);
        for (int j = 0; j < 4; j++) {
            v16bf bfr = ld_fragB(sB, nw * 64 + j * 16, DD, k0, lane);
            acco[j] = wmma_bf16(af, bfr, acco[j]);
        }
    }
    __syncthreads();

    // Spill o to LDS (reuse sB as fp32 obuf, 64x128).
    float* obuf = (float*)sB;
    {
        int l15 = lane & 15, hh = (lane >> 4) & 1;
        for (int j = 0; j < 4; j++) {
            int col = nw * 64 + j * 16 + l15;
            for (int r = 0; r < 8; r++) {
                int row = mw * 16 + r + hh * 8;
                obuf[row * DD + col] = acco[j][r];
            }
        }
    }
    __syncthreads();

    // Epilogue: RMS norm over D, * norm_w, * silu(gate); write bf16 directly.
    for (int rr = 0; rr < 8; rr++) {
        int row = w * 8 + rr;
        float vals[4], ss = 0.0f;
        for (int j = 0; j < 4; j++) {
            vals[j] = obuf[row * DD + lane + 32 * j];
            ss += vals[j] * vals[j];
        }
        for (int off = 16; off; off >>= 1) ss += __shfl_xor(ss, off, 32);
        float rms = rsqrtf(ss * (1.0f / DD) + EPSF);
        size_t grow = rowbase + row;
        for (int j = 0; j < 4; j++) {
            int col = lane + 32 * j;
            float gg = g[grow * HIDD + colbase + col];
            float silu = gg / (1.0f + __expf(-gg));
            O[grow * HIDD + colbase + col] = (bf16_t)(vals[j] * rms * norm_w[col] * silu);
        }
    }
}

// ---------------------------------------------------------------------------
// Host-side orchestration
// ---------------------------------------------------------------------------
extern "C" void kernel_launch(void* const* d_in, const int* in_sizes, int n_in,
                              void* d_out, int out_size, void* d_ws, size_t ws_size,
                              hipStream_t stream) {
    const float* x      = (const float*)d_in[0];
    const float* Wq     = (const float*)d_in[1];
    const float* Wk     = (const float*)d_in[2];
    const float* Wv     = (const float*)d_in[3];
    const float* Wg1    = (const float*)d_in[4];
    const float* Wg2    = (const float*)d_in[5];
    const float* bg2    = (const float*)d_in[6];
    const float* Wgate  = (const float*)d_in[7];
    const float* norm_w = (const float*)d_in[8];
    const float* Wo     = (const float*)d_in[9];
    float* out = (float*)d_out;

    char* ws = (char*)d_ws;
    size_t p = 0;
    const size_t szXbf = (size_t)MM * HIDD * sizeof(bf16_t);
    const size_t szWbf = (size_t)HIDD * HIDD * sizeof(bf16_t);
    const size_t szF   = (size_t)MM * HIDD * sizeof(float);
    const size_t szSst = (size_t)NC * BH * DD * DD * sizeof(float);

    bf16_t* xbf  = (bf16_t*)(ws + p); p += szXbf;
    bf16_t* WqT  = (bf16_t*)(ws + p); p += szWbf;
    bf16_t* WkT  = (bf16_t*)(ws + p); p += szWbf;
    bf16_t* WvT  = (bf16_t*)(ws + p); p += szWbf;
    bf16_t* WgT  = (bf16_t*)(ws + p); p += szWbf;
    bf16_t* WoT  = (bf16_t*)(ws + p); p += szWbf;
    float*  qf   = (float*)(ws + p);  p += szF;
    float*  kf   = (float*)(ws + p);  p += szF;
    float*  vf   = (float*)(ws + p);  p += szF;
    float*  gf   = (float*)(ws + p);  p += szF;
    float*  gkf  = (float*)(ws + p);  p += szF;
    float*  Gcf  = (float*)(ws + p);  p += szF;
    float*  Sst  = (float*)(ws + p);  p += szSst;
    bf16_t* Obf  = (bf16_t*)(ws + p); p += szXbf;

    const size_t nX = (size_t)MM * HIDD;
    const size_t nW = (size_t)HIDD * HIDD;

    // Precision prep: activations/weights -> bf16, weights pre-transposed for WMMA-B.
    f32_to_bf16_kernel<<<(unsigned)((nX + 255) / 256), 256, 0, stream>>>(x, xbf, nX);
    transpose_to_bf16_kernel<<<(unsigned)((nW + 255) / 256), 256, 0, stream>>>(Wq, WqT, HIDD, HIDD);
    transpose_to_bf16_kernel<<<(unsigned)((nW + 255) / 256), 256, 0, stream>>>(Wk, WkT, HIDD, HIDD);
    transpose_to_bf16_kernel<<<(unsigned)((nW + 255) / 256), 256, 0, stream>>>(Wv, WvT, HIDD, HIDD);
    transpose_to_bf16_kernel<<<(unsigned)((nW + 255) / 256), 256, 0, stream>>>(Wgate, WgT, HIDD, HIDD);
    transpose_to_bf16_kernel<<<(unsigned)((nW + 255) / 256), 256, 0, stream>>>(Wo, WoT, HIDD, HIDD);

    // Low-rank gate path (fp32 VALU) + per-chunk cumsum.
    lowrank_gate_kernel<<<MM, 256, 0, stream>>>(x, Wg1, Wg2, bg2, gkf);
    cumsum_kernel<<<BB * NC, 256, 0, stream>>>(gkf, Gcf);

    // Projections via pipelined WMMA GEMM.
    dim3 gg(HIDD / 128, MM / 128);
    gemm_bf16_kernel<<<gg, 256, 0, stream>>>(xbf, WqT, qf, MM, HIDD, HIDD);
    gemm_bf16_kernel<<<gg, 256, 0, stream>>>(xbf, WkT, kf, MM, HIDD, HIDD);
    gemm_bf16_kernel<<<gg, 256, 0, stream>>>(xbf, WvT, vf, MM, HIDD, HIDD);
    gemm_bf16_kernel<<<gg, 256, 0, stream>>>(xbf, WgT, gf, MM, HIDD, HIDD);

    // Inter-chunk state recurrence; final state goes straight into d_out tail.
    state_scan_kernel<<<512, 256, 0, stream>>>(kf, vf, Gcf, Sst, out + nX);

    // Per-chunk WMMA attention output + RMSNorm + SiLU gate, bf16 out.
    dim3 gc(NC, BH);
    chunk_output_kernel<<<gc, 256, 0, stream>>>(qf, kf, vf, Gcf, Sst, gf, norm_w, Obf);

    // Final projection o @ Wo -> d_out.
    gemm_bf16_kernel<<<gg, 256, 0, stream>>>(Obf, WoT, out, MM, HIDD, HIDD);

    (void)in_sizes; (void)n_in; (void)out_size; (void)ws_size;
}
